// IN_CGCNN_Conv_7499012898892
// MI455X (gfx1250) — matrix operand order repaired and checked by reading, not compiled, not run on hardware
//
#include <hip/hip_runtime.h>
#include <hip/hip_bf16.h>

typedef __bf16 bf16;
typedef __attribute__((ext_vector_type(16))) __bf16 v16bf;
typedef __attribute__((ext_vector_type(8)))  __bf16 v8bf;
typedef __attribute__((ext_vector_type(8)))  float  v8f;

#define NG 512          // NUM_GRAPHS

// ---------------------------------------------------------------------------
// CDNA5 WMMA fragment loaders (wave32, 16x16x32 bf16).
// A-matrix 16x32 (MxK), row-major source with leading dim ld:
//   lane L: M = L&15, half = L>>4
//   vector elems 0..7  <-> K = k0 + half*8  + (0..7)     (contiguous)
//   vector elems 8..15 <-> K = k0 + 16 + half*8 + (0..7) (contiguous)
// ---------------------------------------------------------------------------
__device__ inline v16bf load_fragA(const bf16* base, int rowM, int ld, int k0, int lane) {
    int M = rowM + (lane & 15);
    int half = lane >> 4;
    const bf16* p = base + (size_t)M * ld + k0 + half * 8;
    union { v16bf v; v8bf h[2]; } u;
    u.h[0] = *(const v8bf*)(p);
    u.h[1] = *(const v8bf*)(p + 16);
    return u.v;
}

// B-matrix 32x16 (KxN), source is column-major weights Wt[n*128 + k]:
//   lane L: N = colN + (L&15), half = L>>4
//   vector elems 0..7  <-> K = k0 + half*8 + (0..7), elems 8..15 <-> K = k0 + 16 + half*8 + (0..7)
__device__ inline v16bf load_fragB(const bf16* Wt, int colN, int k0, int lane) {
    int n = colN + (lane & 15);
    int half = lane >> 4;
    const bf16* p = Wt + (size_t)n * 128 + k0 + half * 8;
    union { v16bf v; v8bf h[2]; } u;
    u.h[0] = *(const v8bf*)(p);
    u.h[1] = *(const v8bf*)(p + 16);
    return u.v;
}

__device__ inline v8f wmma_bf16(v16bf a, v16bf b, v8f c) {
    return __builtin_amdgcn_wmma_f32_16x16x32_bf16(false, a, false, b, (short)0, c, false, false);
}

__device__ inline float softplus_f(float x) {
    return (x > 20.0f) ? x : log1pf(__expf(x));
}

// ---------------------------------------------------------------------------
// Weight prep: transpose the 3 column-blocks of Wf/Ws [5][384][128] into
// column-major bf16 tiles Wbf[layer][mat][n][k]:
//   mat 0 = Wf rows   0..127 (src)   mat 1 = Wf rows 128..255 (tgt)
//   mat 2 = Ws rows   0..127 (src)   mat 3 = Ws rows 128..255 (tgt)
//   mat 4 = Wf rows 256..383 (edge)  mat 5 = Ws rows 256..383 (edge)
// ---------------------------------------------------------------------------
__global__ void prep_weights(const float* __restrict__ Wf, const float* __restrict__ Ws,
                             bf16* __restrict__ Wbf) {
    int idx = blockIdx.x * 256 + threadIdx.x;
    const int total = 5 * 6 * 128 * 128;
    if (idx >= total) return;
    int k   = idx & 127;
    int n   = (idx >> 7) & 127;
    int mat = (idx >> 14) % 6;
    int layer = idx / (6 * 128 * 128);
    const float* W; int part;
    switch (mat) {
        case 0: W = Wf; part = 0; break;
        case 1: W = Wf; part = 1; break;
        case 2: W = Ws; part = 0; break;
        case 3: W = Ws; part = 1; break;
        case 4: W = Wf; part = 2; break;
        default: W = Ws; part = 2; break;
    }
    float v = W[((size_t)layer * 384 + part * 128 + k) * 128 + n];
    Wbf[idx] = (bf16)v;
}

// x = relu(xin @ Wn + bn), also stage bf16 copy
__global__ void embed_nodes(const float* __restrict__ xin, const float* __restrict__ W,
                            const float* __restrict__ b, float* __restrict__ x,
                            bf16* __restrict__ xb) {
    __shared__ float row[92];
    int n = blockIdx.x;
    int d = threadIdx.x;               // 128 threads
    if (d < 92) row[d] = xin[(size_t)n * 92 + d];
    __syncthreads();
    float acc = b[d];
    #pragma unroll 4
    for (int k = 0; k < 92; ++k) acc = fmaf(row[k], W[k * 128 + d], acc);
    acc = fmaxf(acc, 0.0f);
    x[(size_t)n * 128 + d]  = acc;
    xb[(size_t)n * 128 + d] = (bf16)acc;
}

// e = relu(edge_attr @ We + be) -> bf16 only (e used only as GEMM operand)
__global__ void embed_edges(const float* __restrict__ ein, const float* __restrict__ W,
                            const float* __restrict__ b, bf16* __restrict__ eb) {
    __shared__ float row[41];
    int e = blockIdx.x;
    int d = threadIdx.x;               // 128 threads
    if (d < 41) row[d] = ein[(size_t)e * 41 + d];
    __syncthreads();
    float acc = b[d];
    #pragma unroll
    for (int k = 0; k < 41; ++k) acc = fmaf(row[k], W[k * 128 + d], acc);
    eb[(size_t)e * 128 + d] = (bf16)fmaxf(acc, 0.0f);
}

// Af/Bf/As/Bs = x_bf16 @ Wt[sel]   ([N,128]x[128,128], WMMA bf16)
// grid: (N/64, 4), block: 256 = 8 waves, wave w -> column block 16w.
// B fragments hoisted once, reused across 4 row tiles.
__global__ void node_gemm4(const bf16* __restrict__ xb, const bf16* __restrict__ Wl,
                           float* __restrict__ outs, int N) {
    int lane = threadIdx.x & 31;
    int wave = threadIdx.x >> 5;
    int rowM = blockIdx.x * 64;
    int sel  = blockIdx.y;                       // 0..3 -> mats 0..3
    const bf16* Wt = Wl + (size_t)sel * 128 * 128;
    float* out = outs + (size_t)sel * N * 128;
    int colN = wave * 16;

    v16bf Bt[4];
    #pragma unroll
    for (int kk = 0; kk < 4; ++kk) Bt[kk] = load_fragB(Wt, colN, kk * 32, lane);

    int half = lane >> 4;
    int col  = colN + (lane & 15);
    #pragma unroll
    for (int rt = 0; rt < 4; ++rt) {
        int row0 = rowM + rt * 16;
        v8f acc = {};
        #pragma unroll
        for (int kk = 0; kk < 4; ++kk) {
            v16bf a = load_fragA(xb, row0, 128, kk * 32, lane);
            acc = wmma_bf16(a, Bt[kk], acc);
        }
        #pragma unroll
        for (int r = 0; r < 8; ++r)
            out[(size_t)(row0 + half * 8 + r) * 128 + col] = acc[r];
    }
}

// Fused: Gf/Gs = e_bf16 @ We_{f,s} (WMMA), then
//   gate = sigmoid(Gf + bf + Af[src] + Bf[tgt]) * softplus(Gs + bs + As[src] + Bs[tgt])
//   msg[src] += gate   (global_atomic_add_f32)
// grid: E/64 edge tiles, block 256 = 8 waves (one 16-col block each).
// Both gate matrices' B fragments hoisted (8 frags = 64 VGPRs), reused over
// 4 row tiles of 16 edges. Next tile's e rows prefetched.
__global__ void edge_conv(const bf16* __restrict__ eb, const bf16* __restrict__ Wl,
                          const float* __restrict__ Af, const float* __restrict__ Bf,
                          const float* __restrict__ As, const float* __restrict__ Bs,
                          const int* __restrict__ esrc, const int* __restrict__ etgt,
                          const float* __restrict__ biasF, const float* __restrict__ biasS,
                          float* __restrict__ msg, int E) {
    __shared__ int ssrc[64], stgt[64];
    int lane = threadIdx.x & 31;
    int wave = threadIdx.x >> 5;
    int tileE = blockIdx.x * 64;
    if (threadIdx.x < 64) {
        ssrc[threadIdx.x] = esrc[tileE + threadIdx.x];
        stgt[threadIdx.x] = etgt[tileE + threadIdx.x];
    }
    // prefetch next tile's edge features (16KB, 64B per thread) -> global_prefetch_b8
    if (tileE + 64 < E) {
        const char* nxt = (const char*)(eb + (size_t)(tileE + 64) * 128);
        __builtin_prefetch(nxt + threadIdx.x * 64, 0, 1);
    }
    __syncthreads();

    int colN = wave * 16;
    const bf16* Wtf = Wl + (size_t)4 * 128 * 128;
    const bf16* Wts = Wl + (size_t)5 * 128 * 128;

    v16bf BtF[4], BtS[4];
    #pragma unroll
    for (int kk = 0; kk < 4; ++kk) {
        BtF[kk] = load_fragB(Wtf, colN, kk * 32, lane);
        BtS[kk] = load_fragB(Wts, colN, kk * 32, lane);
    }

    int half = lane >> 4;
    int col  = colN + (lane & 15);
    float bF = biasF[col], bS = biasS[col];

    #pragma unroll
    for (int rt = 0; rt < 4; ++rt) {
        int row0 = tileE + rt * 16;
        v8f accF = {};
        v8f accS = {};
        #pragma unroll
        for (int kk = 0; kk < 4; ++kk) {
            v16bf a = load_fragA(eb, row0, 128, kk * 32, lane);
            accF = wmma_bf16(a, BtF[kk], accF);
            accS = wmma_bf16(a, BtS[kk], accS);
        }
        #pragma unroll
        for (int r = 0; r < 8; ++r) {
            int er = rt * 16 + half * 8 + r;
            int s = ssrc[er], t = stgt[er];
            float f = accF[r] + bF + Af[(size_t)s * 128 + col] + Bf[(size_t)t * 128 + col];
            float g = accS[r] + bS + As[(size_t)s * 128 + col] + Bs[(size_t)t * 128 + col];
            float sig = 1.0f / (1.0f + __expf(-f));
            atomicAdd(&msg[(size_t)s * 128 + col], sig * softplus_f(g));
        }
    }
}

// per-column sum / sumsq of msg [N,128]
__global__ void bn_stats(const float* __restrict__ msg, float* __restrict__ bnbuf, int N) {
    int d = threadIdx.x;               // 128 threads
    float s = 0.0f, s2 = 0.0f;
    for (int n = blockIdx.x; n < N; n += gridDim.x) {
        float v = msg[(size_t)n * 128 + d];
        s += v; s2 += v * v;
    }
    atomicAdd(&bnbuf[d], s);
    atomicAdd(&bnbuf[128 + d], s2);
}

__global__ void bn_finalize(const float* __restrict__ bnbuf, const float* __restrict__ gamma,
                            const float* __restrict__ beta, float* __restrict__ sc, int N) {
    int d = threadIdx.x;               // 128 threads
    float inv_n = 1.0f / (float)N;
    float mu  = bnbuf[d] * inv_n;
    float var = bnbuf[128 + d] * inv_n - mu * mu;
    float inv = rsqrtf(var + 1e-5f);
    float scale = gamma[d] * inv;
    sc[d]       = scale;
    sc[128 + d] = beta[d] - mu * scale;
}

// x = softplus(x + msg*scale + shift); refresh bf16; pool into pooled[graph]
__global__ void node_update(float* __restrict__ x, bf16* __restrict__ xb,
                            const float* __restrict__ msg, const float* __restrict__ sc,
                            const int* __restrict__ nbatch, float* __restrict__ pooled, int N) {
    int idx = blockIdx.x * 256 + threadIdx.x;
    if (idx >= N * 128) return;
    int d = idx & 127;
    int n = idx >> 7;
    float v  = x[idx] + msg[idx] * sc[d] + sc[128 + d];
    float sp = softplus_f(v);
    x[idx]  = sp;
    xb[idx] = (bf16)sp;
    atomicAdd(&pooled[(size_t)nbatch[n] * 128 + d], sp);
}

// y += pooled @ Wg[i] + bg[i]    ([512,128]x[128,128], tiny -> fp32 VALU)
__global__ void pool_gemm(const float* __restrict__ pooled, const float* __restrict__ Wg,
                          const float* __restrict__ bg, float* __restrict__ y) {
    __shared__ float row[128];
    int g = blockIdx.x;
    int d = threadIdx.x;
    row[d] = pooled[(size_t)g * 128 + d];
    __syncthreads();
    float acc = bg[d];
    #pragma unroll 8
    for (int k = 0; k < 128; ++k) acc = fmaf(row[k], Wg[k * 128 + d], acc);
    y[(size_t)g * 128 + d] += acc;
}

// readout MLP 128->64->32->16->1 with relu, one block per graph
__global__ void readout(const float* __restrict__ y,
                        const float* __restrict__ W1, const float* __restrict__ b1,
                        const float* __restrict__ W2, const float* __restrict__ b2,
                        const float* __restrict__ W3, const float* __restrict__ b3,
                        const float* __restrict__ W4, const float* __restrict__ b4,
                        float* __restrict__ out) {
    __shared__ float h0[128], h1[64], h2[32], h3[16];
    int g = blockIdx.x;
    int t = threadIdx.x;               // 128 threads
    h0[t] = y[(size_t)g * 128 + t];
    __syncthreads();
    if (t < 64) {
        float a = b1[t];
        for (int k = 0; k < 128; ++k) a = fmaf(h0[k], W1[k * 64 + t], a);
        h1[t] = fmaxf(a, 0.0f);
    }
    __syncthreads();
    if (t < 32) {
        float a = b2[t];
        for (int k = 0; k < 64; ++k) a = fmaf(h1[k], W2[k * 32 + t], a);
        h2[t] = fmaxf(a, 0.0f);
    }
    __syncthreads();
    if (t < 16) {
        float a = b3[t];
        for (int k = 0; k < 32; ++k) a = fmaf(h2[k], W3[k * 16 + t], a);
        h3[t] = fmaxf(a, 0.0f);
    }
    __syncthreads();
    if (t == 0) {
        float a = b4[0];
        for (int k = 0; k < 16; ++k) a = fmaf(h3[k], W4[k], a);
        out[g] = fmaxf(a, 0.0f);
    }
}

__global__ void zero_f32(float* __restrict__ p, int n) {
    int i = blockIdx.x * 256 + threadIdx.x;
    if (i < n) p[i] = 0.0f;
}

extern "C" void kernel_launch(void* const* d_in, const int* in_sizes, int n_in,
                              void* d_out, int out_size, void* d_ws, size_t ws_size,
                              hipStream_t stream) {
    const float* xin   = (const float*)d_in[0];
    const float* eattr = (const float*)d_in[1];
    const int*   esrc  = (const int*)d_in[2];
    const int*   etgt  = (const int*)d_in[3];
    const int*   nbat  = (const int*)d_in[4];
    const float* Wn = (const float*)d_in[5];  const float* bn = (const float*)d_in[6];
    const float* We = (const float*)d_in[7];  const float* be = (const float*)d_in[8];
    const float* Wf = (const float*)d_in[9];  const float* bfv = (const float*)d_in[10];
    const float* Ws = (const float*)d_in[11]; const float* bsv = (const float*)d_in[12];
    const float* Wg = (const float*)d_in[13]; const float* bg  = (const float*)d_in[14];
    const float* gamma = (const float*)d_in[15]; const float* beta = (const float*)d_in[16];
    const float* W1 = (const float*)d_in[17]; const float* b1 = (const float*)d_in[18];
    const float* W2 = (const float*)d_in[19]; const float* b2 = (const float*)d_in[20];
    const float* W3 = (const float*)d_in[21]; const float* b3 = (const float*)d_in[22];
    const float* W4 = (const float*)d_in[23]; const float* b4 = (const float*)d_in[24];
    float* out = (float*)d_out;

    const int N = in_sizes[0] / 92;    // 50000 (multiple of 16)
    const int E = in_sizes[1] / 41;    // 400000 (multiple of 64)

    // ---- workspace carve-out ----
    char* ws = (char*)d_ws;
    size_t off = 0;
    auto carve = [&](size_t bytes) -> void* {
        void* p = ws + off;
        off += (bytes + 255) & ~(size_t)255;
        return p;
    };
    float* x      = (float*)carve((size_t)N * 128 * 4);
    float* msg    = (float*)carve((size_t)N * 128 * 4);
    float* gath   = (float*)carve((size_t)4 * N * 128 * 4);   // Af,Bf,As,Bs
    bf16*  xb     = (bf16*) carve((size_t)N * 128 * 2);
    bf16*  eb     = (bf16*) carve((size_t)E * 128 * 2);
    bf16*  Wbf    = (bf16*) carve((size_t)5 * 6 * 128 * 128 * 2);
    float* pooled = (float*)carve((size_t)NG * 128 * 4);
    float* yb     = (float*)carve((size_t)NG * 128 * 4);
    float* bnbuf  = (float*)carve(256 * 4);
    float* sc     = (float*)carve(256 * 4);
    float* Afp = gath;
    float* Bfp = gath + (size_t)N * 128;
    float* Asp = gath + (size_t)2 * N * 128;
    float* Bsp = gath + (size_t)3 * N * 128;

    // ---- setup ----
    prep_weights<<<(5 * 6 * 128 * 128 + 255) / 256, 256, 0, stream>>>(Wf, Ws, Wbf);
    embed_nodes<<<N, 128, 0, stream>>>(xin, Wn, bn, x, xb);
    embed_edges<<<E, 128, 0, stream>>>(eattr, We, be, eb);
    zero_f32<<<(NG * 128 + 255) / 256, 256, 0, stream>>>(yb, NG * 128);

    // ---- 5 conv layers ----
    for (int i = 0; i < 5; ++i) {
        zero_f32<<<(N * 128 + 255) / 256, 256, 0, stream>>>(msg, N * 128);
        zero_f32<<<1, 256, 0, stream>>>(bnbuf, 256);
        zero_f32<<<(NG * 128 + 255) / 256, 256, 0, stream>>>(pooled, NG * 128);

        const bf16* Wl = Wbf + (size_t)i * 6 * 128 * 128;
        dim3 gGemm(N / 64, 4);
        node_gemm4<<<gGemm, 256, 0, stream>>>(xb, Wl, gath, N);
        edge_conv<<<E / 64, 256, 0, stream>>>(eb, Wl, Afp, Bfp, Asp, Bsp,
                                              esrc, etgt, bfv + i * 128, bsv + i * 128, msg, E);
        bn_stats<<<256, 128, 0, stream>>>(msg, bnbuf, N);
        bn_finalize<<<1, 128, 0, stream>>>(bnbuf, gamma + i * 128, beta + i * 128, sc, N);
        node_update<<<(N * 128 + 255) / 256, 256, 0, stream>>>(x, xb, msg, sc, nbat, pooled, N);
        pool_gemm<<<NG, 128, 0, stream>>>(pooled, Wg + (size_t)i * 128 * 128, bg + i * 128, yb);
    }

    // ---- readout ----
    readout<<<NG, 128, 0, stream>>>(yb, W1, b1, W2, b2, W3, b3, W4, b4, out);
}
